// FastTransformerDecoder_78494822302314
// MI455X (gfx1250) — matrix operand order repaired and checked by reading, not compiled, not run on hardware
//
#include <hip/hip_runtime.h>
#include <hip/hip_bf16.h>

// ---------------------------------------------------------------------------
// FastTransformer (Performer / FAVOR+ causal linear attention) forward,
// MI455X (gfx1250, wave32). GEMMs via v_wmma_f32_16x16x32_bf16,
// 16x64 per-wave tiles, software-pipelined K loop (double-buffered fragments).
// L=2, H=8, DM=512, DFF=2048, DH=M=64, B=1, S=2048.
// ---------------------------------------------------------------------------

#define S_LEN 2048
#define DM    512
#define DFF   2048
#define NHEAD 8
#define DH    64
#define NFEAT 64
#define NCHUNK 32   // S / 64
#define CHUNK  64

typedef __attribute__((ext_vector_type(16))) __bf16 v16bf;
typedef __attribute__((ext_vector_type(8)))  __bf16 v8bf;
typedef __attribute__((ext_vector_type(8)))  float  v8f;

// Build a 16-elem bf16 fragment from two contiguous 8-elem (16B) loads.
__device__ __forceinline__ v16bf load_frag16(const __bf16* p, int gap) {
  v8bf lo = *(const v8bf*)(p);
  v8bf hi = *(const v8bf*)(p + gap);
  return __builtin_shufflevector(lo, hi, 0,1,2,3,4,5,6,7,8,9,10,11,12,13,14,15);
}

// ---------------------------------------------------------------------------
// fp32 -> bf16 elementwise convert (only used for the layer-0 input)
// ---------------------------------------------------------------------------
__global__ void k_conv_bf16(const float* __restrict__ src, __bf16* __restrict__ dst, int n) {
  int i = blockIdx.x * blockDim.x + threadIdx.x;
  if (i < n) dst[i] = (__bf16)src[i];
}

// ---------------------------------------------------------------------------
// W[K][N] fp32 -> Wt[N][K] bf16 (so WMMA B fragment loads are contiguous-K)
// block (32,8), grid (N/32, K/32)
// ---------------------------------------------------------------------------
__global__ void k_transpose_bf16(const float* __restrict__ W, __bf16* __restrict__ Wt,
                                 int K, int N) {
  __shared__ float tile[32][33];
  int nx = blockIdx.x * 32 + threadIdx.x;
  int k0 = blockIdx.y * 32;
  for (int j = 0; j < 32; j += 8)
    tile[threadIdx.y + j][threadIdx.x] = W[(size_t)(k0 + threadIdx.y + j) * N + nx];
  __syncthreads();
  int n0 = blockIdx.x * 32;
  int kx = k0 + threadIdx.x;
  for (int j = 0; j < 32; j += 8)
    Wt[(size_t)(n0 + threadIdx.y + j) * K + kx] = (__bf16)tile[threadIdx.x][threadIdx.y + j];
}

// ---------------------------------------------------------------------------
// C[M,N] = A[M,K] (bf16) x Bt[N,K] (bf16) + bias (+resid)(+relu)
// Optional outputs: C (f32, may be null), Cb (bf16 copy, may be null).
// wave32: 16x64 tile per wave (4 accumulators, A fragment reused), 8 waves
// per block covering 128x64. grid (N/64, M/128), block 256.
// K loop is explicitly double-buffered: next-step fragment loads are issued
// before the current step's WMMAs so loads overlap matrix ops.
//
// VGPR layouts used here (contiguous k order, same for A and B; C per ISA):
//   A 16x32 bf16 : lane m (0-15) holds K {0..15}; lane m+16 holds
//                  K {16..31}  -> two contiguous 8-elem chunks.
//   B 32x16 bf16 : lane n (0-15) holds K {0..15}; lane n+16 holds K {16..31}
//                  of column n -> 16 contiguous elems of Bt row n.
//   C 16x16 f32  : lane<16: M=v, N=lane; lane>=16: M=8+v, N=lane-16.
// ---------------------------------------------------------------------------
__global__ __launch_bounds__(256)
void k_gemm_bf16(const __bf16* __restrict__ A, const __bf16* __restrict__ Bt,
                 const float* __restrict__ bias, const float* __restrict__ resid,
                 float* __restrict__ C, __bf16* __restrict__ Cb,
                 int M, int N, int K, int relu) {
  int wave = threadIdx.x >> 5;
  int lane = threadIdx.x & 31;
  int tileM = blockIdx.y * 128 + wave * 16;
  int tileN = blockIdx.x * 64;
  int lo = lane & 15;
  int hi = lane >> 4;           // 0 or 1

  const __bf16* arow = A  + (size_t)(tileM + lo) * K + hi * 16;
  const __bf16* brow = Bt + (size_t)(tileN + lo) * K + hi * 16;
  const size_t bstr = (size_t)16 * K;   // next N-tile of Bt

  v8f acc0 = {}, acc1 = {}, acc2 = {}, acc3 = {};

  // prologue: fragments for k0 = 0
  v16bf av  = load_frag16(arow, 8);
  v16bf bv0 = load_frag16(brow + 0 * bstr, 8);
  v16bf bv1 = load_frag16(brow + 1 * bstr, 8);
  v16bf bv2 = load_frag16(brow + 2 * bstr, 8);
  v16bf bv3 = load_frag16(brow + 3 * bstr, 8);

  for (int k0 = 0; k0 < K - 32; k0 += 32) {
    int kn = k0 + 32;
    // issue next-step loads before consuming current fragments
    v16bf an  = load_frag16(arow + kn, 8);
    v16bf bn0 = load_frag16(brow + 0 * bstr + kn, 8);
    v16bf bn1 = load_frag16(brow + 1 * bstr + kn, 8);
    v16bf bn2 = load_frag16(brow + 2 * bstr + kn, 8);
    v16bf bn3 = load_frag16(brow + 3 * bstr + kn, 8);

    acc0 = __builtin_amdgcn_wmma_f32_16x16x32_bf16(false, av, false, bv0, (short)0, acc0, false, false);
    acc1 = __builtin_amdgcn_wmma_f32_16x16x32_bf16(false, av, false, bv1, (short)0, acc1, false, false);
    acc2 = __builtin_amdgcn_wmma_f32_16x16x32_bf16(false, av, false, bv2, (short)0, acc2, false, false);
    acc3 = __builtin_amdgcn_wmma_f32_16x16x32_bf16(false, av, false, bv3, (short)0, acc3, false, false);

    av = an; bv0 = bn0; bv1 = bn1; bv2 = bn2; bv3 = bn3;
  }
  // epilogue step
  acc0 = __builtin_amdgcn_wmma_f32_16x16x32_bf16(false, av, false, bv0, (short)0, acc0, false, false);
  acc1 = __builtin_amdgcn_wmma_f32_16x16x32_bf16(false, av, false, bv1, (short)0, acc1, false, false);
  acc2 = __builtin_amdgcn_wmma_f32_16x16x32_bf16(false, av, false, bv2, (short)0, acc2, false, false);
  acc3 = __builtin_amdgcn_wmma_f32_16x16x32_bf16(false, av, false, bv3, (short)0, acc3, false, false);

  v8f accs[4] = {acc0, acc1, acc2, acc3};
  #pragma unroll
  for (int nt = 0; nt < 4; nt++) {
    int ccol = tileN + nt * 16 + lo;
    float bb = bias[ccol];
    #pragma unroll
    for (int v = 0; v < 8; v++) {
      int crow = tileM + hi * 8 + v;
      float val = accs[nt][v] + bb;
      if (resid) val += resid[(size_t)crow * N + ccol];
      if (relu)  val = fmaxf(val, 0.f);
      if (C)  C [(size_t)crow * N + ccol] = val;
      if (Cb) Cb[(size_t)crow * N + ccol] = (__bf16)val;
    }
  }
}

// ---------------------------------------------------------------------------
// FAVOR+ feature map: phi(x) = exp(xs @ omega - 0.5*||xs||^2) * M^-0.5,
// xs = x * DH^-0.25. grid = S*H, block = 64 (thread = feature m).
// ---------------------------------------------------------------------------
__global__ void k_phi(const float* __restrict__ x, const float* __restrict__ omega,
                      float* __restrict__ ph) {
  int s = blockIdx.x >> 3;
  int h = blockIdx.x & 7;
  int m = threadIdx.x;
  __shared__ float xs[64];
  __shared__ float red[64];
  float xv = x[(size_t)s * DM + h * DH + m] * 0.3535533905932738f;  // 64^-0.25
  xs[m] = xv;
  red[m] = xv * xv;
  __syncthreads();
  for (int off = 32; off > 0; off >>= 1) {
    if (m < off) red[m] += red[m + off];
    __syncthreads();
  }
  float nrm = red[0];
  float u = 0.f;
  for (int d = 0; d < DH; d++) u += xs[d] * omega[d * NFEAT + m];
  ph[(size_t)s * DM + h * NFEAT + m] = expf(u - 0.5f * nrm) * 0.125f;  // 64^-0.5
}

// ---------------------------------------------------------------------------
// Per-chunk KV_c[m][d] = sum_i pk[i][m]*v[i][d], Ksum_c[m] = sum_i pk[i][m]
// grid = H*NCHUNK, block = 256.
// ---------------------------------------------------------------------------
__global__ void k_chunk_kv(const float* __restrict__ pk, const float* __restrict__ v,
                           float* __restrict__ KVc, float* __restrict__ Ksum) {
  __shared__ float spk[CHUNK][NFEAT];
  __shared__ float sv[CHUNK][DH];
  int h = blockIdx.x >> 5;
  int c = blockIdx.x & 31;
  int t = threadIdx.x;
  int s0 = c * CHUNK;
  for (int idx = t; idx < CHUNK * 64; idx += 256) {
    int i = idx >> 6, e = idx & 63;
    spk[i][e] = pk[(size_t)(s0 + i) * DM + h * NFEAT + e];
    sv[i][e]  = v [(size_t)(s0 + i) * DM + h * DH + e];
  }
  __syncthreads();
  float* out = KVc + (size_t)(h * NCHUNK + c) * (NFEAT * DH);
  for (int idx = t; idx < NFEAT * DH; idx += 256) {
    int m = idx >> 6, d = idx & 63;
    float acc = 0.f;
    for (int i = 0; i < CHUNK; i++) acc += spk[i][m] * sv[i][d];
    out[idx] = acc;
  }
  if (t < NFEAT) {
    float acc = 0.f;
    for (int i = 0; i < CHUNK; i++) acc += spk[i][t];
    Ksum[(size_t)(h * NCHUNK + c) * NFEAT + t] = acc;
  }
}

// ---------------------------------------------------------------------------
// Exclusive prefix sum across chunks of KV_c and Ksum_c. grid = H, block 256.
// ---------------------------------------------------------------------------
__global__ void k_prefix(const float* __restrict__ KVc, const float* __restrict__ Ksum,
                         float* __restrict__ KVpre, float* __restrict__ Kpre) {
  int h = blockIdx.x;
  int t = threadIdx.x;
  float acc[16];
  #pragma unroll
  for (int j = 0; j < 16; j++) acc[j] = 0.f;
  for (int c = 0; c < NCHUNK; c++) {
    size_t base = (size_t)(h * NCHUNK + c) * (NFEAT * DH);
    #pragma unroll
    for (int j = 0; j < 16; j++) {
      int idx = t + j * 256;
      KVpre[base + idx] = acc[j];
      acc[j] += KVc[base + idx];
    }
  }
  if (t < NFEAT) {
    float a2 = 0.f;
    for (int c = 0; c < NCHUNK; c++) {
      size_t base = (size_t)(h * NCHUNK + c) * NFEAT;
      Kpre[base + t] = a2;
      a2 += Ksum[base + t];
    }
  }
}

// ---------------------------------------------------------------------------
// Chunked causal linear attention output (bf16 out, feeds Wo projection):
//   a[i] = (sum_{j<=i in chunk} (pq_i . pk_j) v_j  +  pq_i @ KVpre) / den
//   den  = sum_{j<=i} (pq_i . pk_j) + pq_i . Kpre + eps
// grid = H*NCHUNK, block 256. LDS: pq,pk,S tiles (48.25 KB).
// ---------------------------------------------------------------------------
__global__ __launch_bounds__(256)
void k_attn_out(const float* __restrict__ pq, const float* __restrict__ pk,
                const float* __restrict__ v, const float* __restrict__ KVpre,
                const float* __restrict__ Kpre, __bf16* __restrict__ ab) {
  __shared__ float spq[CHUNK][NFEAT];
  __shared__ float spk[CHUNK][NFEAT];
  __shared__ float sS[CHUNK][CHUNK];
  __shared__ float sden[CHUNK];
  int h = blockIdx.x >> 5;
  int c = blockIdx.x & 31;
  int t = threadIdx.x;
  int s0 = c * CHUNK;
  for (int idx = t; idx < CHUNK * NFEAT; idx += 256) {
    int i = idx >> 6, e = idx & 63;
    spq[i][e] = pq[(size_t)(s0 + i) * DM + h * NFEAT + e];
    spk[i][e] = pk[(size_t)(s0 + i) * DM + h * NFEAT + e];
  }
  __syncthreads();
  for (int idx = t; idx < CHUNK * CHUNK; idx += 256) {
    int i = idx >> 6, j = idx & 63;
    float acc = 0.f;
    if (j <= i) {
      for (int m = 0; m < NFEAT; m++) acc += spq[i][m] * spk[j][m];
    }
    sS[i][j] = acc;
  }
  __syncthreads();
  const float* kpre = Kpre + (size_t)(h * NCHUNK + c) * NFEAT;
  if (t < CHUNK) {
    float den = 0.f;
    for (int j = 0; j < CHUNK; j++) den += sS[t][j];
    for (int m = 0; m < NFEAT; m++) den += spq[t][m] * kpre[m];
    sden[t] = den + 1e-6f;
  }
  __syncthreads();
  const float* kv = KVpre + (size_t)(h * NCHUNK + c) * (NFEAT * DH);
  for (int idx = t; idx < CHUNK * DH; idx += 256) {
    int i = idx >> 6, d = idx & 63;
    float acc = 0.f;
    for (int j = 0; j < CHUNK; j++)
      acc += sS[i][j] * v[(size_t)(s0 + j) * DM + h * DH + d];
    for (int m = 0; m < NFEAT; m++)
      acc += spq[i][m] * kv[m * DH + d];
    ab[(size_t)(s0 + i) * DM + h * DH + d] = (__bf16)(acc / sden[i]);
  }
}

// ---------------------------------------------------------------------------
// LayerNorm over last dim (512). Optional bf16 secondary output.
// grid = S, block 256 (2 elems/thread).
// ---------------------------------------------------------------------------
__global__ void k_layernorm(const float* __restrict__ x, const float* __restrict__ g,
                            const float* __restrict__ b, float* __restrict__ y,
                            __bf16* __restrict__ yb) {
  __shared__ float red[256];
  int s = blockIdx.x;
  int t = threadIdx.x;
  float x0 = x[(size_t)s * DM + t];
  float x1 = x[(size_t)s * DM + 256 + t];
  red[t] = x0 + x1;
  __syncthreads();
  for (int off = 128; off > 0; off >>= 1) {
    if (t < off) red[t] += red[t + off];
    __syncthreads();
  }
  float mu = red[0] * (1.f / DM);
  __syncthreads();
  float d0 = x0 - mu, d1 = x1 - mu;
  red[t] = d0 * d0 + d1 * d1;
  __syncthreads();
  for (int off = 128; off > 0; off >>= 1) {
    if (t < off) red[t] += red[t + off];
    __syncthreads();
  }
  float rs = rsqrtf(red[0] * (1.f / DM) + 1e-5f);
  float y0 = d0 * rs * g[t] + b[t];
  float y1 = d1 * rs * g[256 + t] + b[256 + t];
  y[(size_t)s * DM + t]       = y0;
  y[(size_t)s * DM + 256 + t] = y1;
  if (yb) {
    yb[(size_t)s * DM + t]       = (__bf16)y0;
    yb[(size_t)s * DM + 256 + t] = (__bf16)y1;
  }
}

// ---------------------------------------------------------------------------
// Host orchestration
// ---------------------------------------------------------------------------
extern "C" void kernel_launch(void* const* d_in, const int* in_sizes, int n_in,
                              void* d_out, int out_size, void* d_ws, size_t ws_size,
                              hipStream_t stream) {
  const float* X   = (const float*)d_in[0];
  const float* Wq  = (const float*)d_in[1];
  const float* Wk  = (const float*)d_in[2];
  const float* Wv  = (const float*)d_in[3];
  const float* Wo  = (const float*)d_in[4];
  const float* W1  = (const float*)d_in[5];
  const float* W2  = (const float*)d_in[6];
  const float* bq  = (const float*)d_in[7];
  const float* bk  = (const float*)d_in[8];
  const float* bv  = (const float*)d_in[9];
  const float* bo  = (const float*)d_in[10];
  const float* b1  = (const float*)d_in[11];
  const float* b2  = (const float*)d_in[12];
  const float* g1  = (const float*)d_in[13];
  const float* be1 = (const float*)d_in[14];
  const float* g2  = (const float*)d_in[15];
  const float* be2 = (const float*)d_in[16];
  const float* omg = (const float*)d_in[17];

  char* base = (char*)d_ws;
  size_t off = 0;
  auto alloc = [&](size_t bytes) -> void* {
    off = (off + 255) & ~(size_t)255;
    void* p = base + off;
    off += bytes;
    return p;
  };

  const size_t SD = (size_t)S_LEN * DM;   // 1M elems
  const size_t SF = (size_t)S_LEN * DFF;  // 4M elems

  __bf16* WqT = (__bf16*)alloc((size_t)DM * DM * 2);
  __bf16* WkT = (__bf16*)alloc((size_t)DM * DM * 2);
  __bf16* WvT = (__bf16*)alloc((size_t)DM * DM * 2);
  __bf16* WoT = (__bf16*)alloc((size_t)DM * DM * 2);
  __bf16* W1T = (__bf16*)alloc((size_t)DM * DFF * 2);
  __bf16* W2T = (__bf16*)alloc((size_t)DFF * DM * 2);
  __bf16* xb  = (__bf16*)alloc(SD * 2);
  float*  q   = (float*)alloc(SD * 4);
  float*  k   = (float*)alloc(SD * 4);
  float*  v   = (float*)alloc(SD * 4);
  float*  pq  = (float*)alloc(SD * 4);
  float*  pk  = (float*)alloc(SD * 4);
  float*  KVc  = (float*)alloc((size_t)NHEAD * NCHUNK * NFEAT * DH * 4);
  float*  KVpre= (float*)alloc((size_t)NHEAD * NCHUNK * NFEAT * DH * 4);
  float*  Ksum = (float*)alloc((size_t)NHEAD * NCHUNK * NFEAT * 4);
  float*  Kpre = (float*)alloc((size_t)NHEAD * NCHUNK * NFEAT * 4);
  __bf16* ab   = (__bf16*)alloc(SD * 2);
  float*  t0   = (float*)alloc(SD * 4);
  float*  x1   = (float*)alloc(SD * 4);
  __bf16* x1b  = (__bf16*)alloc(SD * 2);
  __bf16* h1b  = (__bf16*)alloc(SF * 2);
  float*  t2   = (float*)alloc(SD * 4);
  float*  cur  = (float*)alloc(SD * 4);
  (void)ws_size; (void)n_in; (void)in_sizes; (void)out_size;

  dim3 blkT(32, 8);
  const int L = 2;

  // layer-0 input -> bf16 (later layers get bf16 from fused LN output)
  k_conv_bf16<<<SD / 256, 256, 0, stream>>>(X, xb, (int)SD);

  for (int l = 0; l < L; l++) {
    const float* src = (l == 0) ? X : cur;
    float* lnout = (l == L - 1) ? (float*)d_out : cur;
    __bf16* lnoutb = (l == L - 1) ? nullptr : xb;   // bf16 input for next layer

    // weight convert+transpose (per layer)
    k_transpose_bf16<<<dim3(DM / 32, DM / 32), blkT, 0, stream>>>(Wq + (size_t)l * DM * DM, WqT, DM, DM);
    k_transpose_bf16<<<dim3(DM / 32, DM / 32), blkT, 0, stream>>>(Wk + (size_t)l * DM * DM, WkT, DM, DM);
    k_transpose_bf16<<<dim3(DM / 32, DM / 32), blkT, 0, stream>>>(Wv + (size_t)l * DM * DM, WvT, DM, DM);
    k_transpose_bf16<<<dim3(DM / 32, DM / 32), blkT, 0, stream>>>(Wo + (size_t)l * DM * DM, WoT, DM, DM);
    k_transpose_bf16<<<dim3(DFF / 32, DM / 32), blkT, 0, stream>>>(W1 + (size_t)l * DM * DFF, W1T, DM, DFF);
    k_transpose_bf16<<<dim3(DM / 32, DFF / 32), blkT, 0, stream>>>(W2 + (size_t)l * DFF * DM, W2T, DFF, DM);

    // QKV projections (f32 out)
    k_gemm_bf16<<<dim3(DM / 64, S_LEN / 128), 256, 0, stream>>>(xb, WqT, bq + l * DM, nullptr, q, nullptr, S_LEN, DM, DM, 0);
    k_gemm_bf16<<<dim3(DM / 64, S_LEN / 128), 256, 0, stream>>>(xb, WkT, bk + l * DM, nullptr, k, nullptr, S_LEN, DM, DM, 0);
    k_gemm_bf16<<<dim3(DM / 64, S_LEN / 128), 256, 0, stream>>>(xb, WvT, bv + l * DM, nullptr, v, nullptr, S_LEN, DM, DM, 0);

    // FAVOR+ features
    k_phi<<<S_LEN * NHEAD, 64, 0, stream>>>(q, omg + (size_t)l * DH * NFEAT, pq);
    k_phi<<<S_LEN * NHEAD, 64, 0, stream>>>(k, omg + (size_t)l * DH * NFEAT, pk);

    // chunked causal linear attention (emits bf16 for Wo projection)
    k_chunk_kv<<<NHEAD * NCHUNK, 256, 0, stream>>>(pk, v, KVc, Ksum);
    k_prefix<<<NHEAD, 256, 0, stream>>>(KVc, Ksum, KVpre, Kpre);
    k_attn_out<<<NHEAD * NCHUNK, 256, 0, stream>>>(pq, pk, v, KVpre, Kpre, ab);

    // output projection + residual, LN1 (emits f32 residual + bf16 FFN input)
    k_gemm_bf16<<<dim3(DM / 64, S_LEN / 128), 256, 0, stream>>>(ab, WoT, bo + l * DM, src, t0, nullptr, S_LEN, DM, DM, 0);
    k_layernorm<<<S_LEN, 256, 0, stream>>>(t0, g1 + l * DM, be1 + l * DM, x1, x1b);

    // FFN: relu GEMM emits bf16 only; second GEMM adds residual; LN2 fuses
    // the bf16 conversion for the next layer's input.
    k_gemm_bf16<<<dim3(DFF / 64, S_LEN / 128), 256, 0, stream>>>(x1b, W1T, b1 + l * DFF, nullptr, nullptr, h1b, S_LEN, DFF, DM, 1);
    k_gemm_bf16<<<dim3(DM / 64, S_LEN / 128), 256, 0, stream>>>(h1b, W2T, b2 + l * DM, x1, t2, nullptr, S_LEN, DM, DFF, 0);
    k_layernorm<<<S_LEN, 256, 0, stream>>>(t2, g2 + l * DM, be2 + l * DM, lnout, lnoutb);
  }
}
